// PlayerInteractionGCN_46583215292450
// MI455X (gfx1250) — matrix operand order repaired and checked by reading, not compiled, not run on hardware
//
#include <hip/hip_runtime.h>

#define N_NODES 100000
#define N_EDGES 1600000
#define IN_C    128
#define HID_C   128
#define OUT_C   64
#define N_STRIPS (N_NODES / 16)   // 6250, exact

typedef __attribute__((ext_vector_type(16))) __bf16 v16bf;
typedef __attribute__((ext_vector_type(8)))  float  v8f;
typedef __attribute__((ext_vector_type(4)))  unsigned int v4u;
typedef __attribute__((ext_vector_type(8)))  int v8i;
typedef __attribute__((ext_vector_type(4)))  int v4i;

__device__ __forceinline__ void atomAddF(float* p, float v) {
    unsafeAtomicAdd(p, v);      // global_atomic_add_f32, resolves in L2
}

// ---------------- degree / normalization ----------------

__global__ void k_deg_init(float* deg) {
    int i = blockIdx.x * blockDim.x + threadIdx.x;
    if (i < N_NODES) deg[i] = 1.0f;               // self-loop contribution
}

__global__ void k_deg_edges(const int* __restrict__ dst, float* __restrict__ deg) {
    int e = blockIdx.x * blockDim.x + threadIdx.x;
    if (e < N_EDGES) atomAddF(&deg[dst[e]], 1.0f);
}

__global__ void k_inv_sqrt(float* deg_inv) {
    int i = blockIdx.x * blockDim.x + threadIdx.x;
    if (i < N_NODES) deg_inv[i] = rsqrtf(deg_inv[i]);   // deg >= 1 always
}

// ---------------- W pre-pack: f32 row-major -> bf16 WMMA B-fragment order ----
// Fragment order: elem((t,u,lane,v)) at flat ((t*U + u)*32 + lane)*16 + v
//   holds W[(32t + 8*(lane/16) + (v%8) + 16*(v/8))][16u + (lane%16)]  (ISA 32x16 16-bit B layout)
template <int K, int N>
__global__ void k_pack_W(const float* __restrict__ W, __bf16* __restrict__ Wfrag) {
    constexpr int U = N / 16;
    int idx = blockIdx.x * blockDim.x + threadIdx.x;
    if (idx >= K * N) return;
    int v    = idx & 15;
    int lane = (idx >> 4) & 31;
    int u    = (idx >> 9) % U;
    int t    = (idx >> 9) / U;
    int k = 32 * t + 8 * (lane >> 4) + (v & 7) + 16 * (v >> 3);
    int n = 16 * u + (lane & 15);
    Wfrag[idx] = (__bf16)W[k * N + n];
}

// ---------------- TDM flat copy: Wfrag (global, bf16) -> LDS ----------------
// 1-D descriptor: data_size=2B, tensor_dim0 = tile_dim0 = nelems, dims 1+ unused.
// amdgpu-toolchain (clang-23) form: 6 args, extra int32x8 group before cpol.
__device__ __forceinline__ void tdm_copy_to_lds(const __bf16* gsrc, unsigned lds_off,
                                                unsigned nelems) {
    unsigned long long ga = (unsigned long long)(uintptr_t)gsrc;
    v4u g0;
    g0[0] = 1u;                                           // count=1, user descriptor
    g0[1] = lds_off;                                      // lds_addr (bytes)
    g0[2] = (unsigned)ga;                                 // global_addr[31:0]
    g0[3] = ((unsigned)(ga >> 32) & 0x01FFFFFFu) | (2u << 30); // addr[56:32], type=2
    v8i g1;
    g1[0] = (int)(1u << 16);                              // data_size=1 (2 bytes)
    g1[1] = (int)((nelems & 0xFFFFu) << 16);              // tensor_dim0[15:0] @ bits63:48
    g1[2] = (int)((nelems >> 16) & 0xFFFFu);              // tensor_dim0[31:16] @ bits79:64
    g1[3] = (int)((nelems & 0xFFFFu) << 16);              // tile_dim0 @ bits127:112
    g1[4] = 0;                                            // tile_dim1/2 unused
    g1[5] = (int)nelems;                                  // tensor_dim0_stride[31:0]
    g1[6] = 0;
    g1[7] = 0;
    v4i gz4 = {0, 0, 0, 0};
    v8i gz8 = {0, 0, 0, 0, 0, 0, 0, 0};
    __builtin_amdgcn_tensor_load_to_lds(g0, g1, gz4, gz4, gz8, 0);
    __builtin_amdgcn_s_wait_tensorcnt(0);
}

// ---------------- WMMA GEMM: H[16*strips x N] = X @ W (frag-packed) ----------
// Block = 8 waves; wave owns one 16-row strip x all N cols (U=N/16 tiles).
// A frag (16-bit 16x32): lane -> row M=lane%16, half=lane/16,
//   bf16 j(0..7): K=kc+8*half+j ; j(8..15): K=kc+16+8*half+(j-8)
// C/D (f32 16x16): VGPR r, lane -> M=r+8*half, N=lane%16
template <int K, int N, bool RELU>
__global__ void k_gemm_wmma(const float* __restrict__ X,
                            const __bf16* __restrict__ Wfrag,
                            float* __restrict__ H) {
    constexpr int T = K / 32, U = N / 16;
    __shared__ __align__(32) __bf16 sW[K * N];            // fragment-ordered

    if ((threadIdx.x >> 5) == 0) {                        // wave 0 issues the DMA
        tdm_copy_to_lds(Wfrag, (unsigned)(uintptr_t)&sW[0], (unsigned)(K * N));
    }
    __syncthreads();

    const int lane  = threadIdx.x & 31;
    const int mn    = lane & 15;
    const int half  = lane >> 4;
    const int strip = blockIdx.x * 8 + (threadIdx.x >> 5);

    if (strip < N_STRIPS) {                               // wave-uniform guard
        const float* rowp = X + ((size_t)strip * 16 + mn) * K + 8 * half;

        v8f acc[U] = {};
#pragma unroll
        for (int t = 0; t < T; ++t) {
            const int kc = 32 * t;
            v16bf a;
#pragma unroll
            for (int j = 0; j < 8; ++j) {
                float f0 = rowp[kc + j];
                float f1 = rowp[kc + 16 + j];
                if (RELU) { f0 = fmaxf(f0, 0.0f); f1 = fmaxf(f1, 0.0f); }
                a[j]     = (__bf16)f0;
                a[8 + j] = (__bf16)f1;
            }
#pragma unroll
            for (int u = 0; u < U; ++u) {
                v16bf b = *(const v16bf*)&sW[((t * U + u) * 32 + lane) * 16];
                acc[u] = __builtin_amdgcn_wmma_f32_16x16x32_bf16(
                    false, a, false, b, (short)0, acc[u], false, false);
            }
        }

        float* op = H + ((size_t)strip * 16 + 8 * half) * N + mn;
#pragma unroll
        for (int u = 0; u < U; ++u)
#pragma unroll
            for (int r = 0; r < 8; ++r)
                op[(size_t)r * N + u * 16] = acc[u][r];
    }
}

// ---------------- aggregation ----------------

// out[i][c] = bias[c] + h[i][c] * inv[i]^2  (self-loop + bias; fully overwrites out)
template <int C, int SHIFT>
__global__ void k_agg_init(const float* __restrict__ h,
                           const float* __restrict__ inv,
                           const float* __restrict__ bias,
                           float* __restrict__ out) {
    size_t idx = (size_t)blockIdx.x * blockDim.x + threadIdx.x;
    if (idx >= (size_t)N_NODES * C) return;
    int i = (int)(idx >> SHIFT);
    int c = (int)(idx & (C - 1));
    float w = inv[i];
    out[idx] = bias[c] + h[idx] * w * w;
}

// one wave per edge, lane covers 4 channels (128ch layer)
__global__ void k_edge_agg_128(const float* __restrict__ h,
                               const float* __restrict__ inv,
                               const int* __restrict__ src,
                               const int* __restrict__ dst,
                               float* __restrict__ out) {
    int e = blockIdx.x * 8 + (threadIdx.x >> 5);
    if (e >= N_EDGES) return;
    int lane = threadIdx.x & 31;
    int s = src[e], d = dst[e];
    float w = inv[s] * inv[d];
    float4 v = ((const float4*)(h + (size_t)s * 128))[lane];
    float* op = out + (size_t)d * 128 + lane * 4;
    atomAddF(op + 0, v.x * w);
    atomAddF(op + 1, v.y * w);
    atomAddF(op + 2, v.z * w);
    atomAddF(op + 3, v.w * w);
}

// one wave per edge, lane covers 2 channels (64ch layer)
__global__ void k_edge_agg_64(const float* __restrict__ h,
                              const float* __restrict__ inv,
                              const int* __restrict__ src,
                              const int* __restrict__ dst,
                              float* __restrict__ out) {
    int e = blockIdx.x * 8 + (threadIdx.x >> 5);
    if (e >= N_EDGES) return;
    int lane = threadIdx.x & 31;
    int s = src[e], d = dst[e];
    float w = inv[s] * inv[d];
    float2 v = ((const float2*)(h + (size_t)s * 64))[lane];
    float* op = out + (size_t)d * 64 + lane * 2;
    atomAddF(op + 0, v.x * w);
    atomAddF(op + 1, v.y * w);
}

// ---------------- launch ----------------

extern "C" void kernel_launch(void* const* d_in, const int* in_sizes, int n_in,
                              void* d_out, int out_size, void* d_ws, size_t ws_size,
                              hipStream_t stream) {
    const float* x  = (const float*)d_in[0];
    const int*   ei = (const int*)d_in[1];
    const float* W1 = (const float*)d_in[2];
    const float* b1 = (const float*)d_in[3];
    const float* W2 = (const float*)d_in[4];
    const float* b2 = (const float*)d_in[5];
    float* out = (float*)d_out;

    const int* src = ei;
    const int* dst = ei + N_EDGES;

    char*   ws     = (char*)d_ws;
    float*  inv    = (float*)ws;                           // N floats (also deg)
    __bf16* Wfrag1 = (__bf16*)(ws + 401408);               // 128*128 bf16 = 32KB
    __bf16* Wfrag2 = (__bf16*)(ws + 434176);               // 128*64  bf16 = 16KB
    float*  bufA   = (float*)(ws + (1u << 19));            // N*128 f32 (h1, then h2)
    float*  bufB   = (float*)(ws + (1u << 19) + (size_t)N_NODES * HID_C * 4); // out1

    // 1) degree -> inv_sqrt ; pack weights to fragment order
    k_deg_init<<<(N_NODES + 255) / 256, 256, 0, stream>>>(inv);
    k_deg_edges<<<(N_EDGES + 255) / 256, 256, 0, stream>>>(dst, inv);
    k_inv_sqrt<<<(N_NODES + 255) / 256, 256, 0, stream>>>(inv);
    k_pack_W<IN_C, HID_C><<<(IN_C * HID_C) / 256, 256, 0, stream>>>(W1, Wfrag1);
    k_pack_W<HID_C, OUT_C><<<(HID_C * OUT_C) / 256, 256, 0, stream>>>(W2, Wfrag2);

    const int gemm_blocks = (N_STRIPS + 7) / 8;            // 782

    // 2) h1 = x @ W1  (TDM-staged B, bf16 WMMA -> f32)
    k_gemm_wmma<IN_C, HID_C, false><<<gemm_blocks, 256, 0, stream>>>(x, Wfrag1, bufA);

    // 3) out1 = b1 + self-loop + scatter(edges)
    k_agg_init<HID_C, 7><<<(N_NODES * HID_C + 255) / 256, 256, 0, stream>>>(bufA, inv, b1, bufB);
    k_edge_agg_128<<<N_EDGES / 8, 256, 0, stream>>>(bufA, inv, src, dst, bufB);

    // 4) h2 = relu(out1) @ W2  (ReLU fused into A-fragment load)
    k_gemm_wmma<HID_C, OUT_C, true><<<gemm_blocks, 256, 0, stream>>>(bufB, Wfrag2, bufA);

    // 5) out = b2 + self-loop + scatter(edges)
    k_agg_init<OUT_C, 6><<<(N_NODES * OUT_C + 255) / 256, 256, 0, stream>>>(bufA, inv, b2, out);
    k_edge_agg_64<<<N_EDGES / 8, 256, 0, stream>>>(bufA, inv, src, dst, out);
}
